// DeepQNetwork2_78735340470674
// MI455X (gfx1250) — matrix operand order repaired and checked by reading, not compiled, not run on hardware
//
#include <hip/hip_runtime.h>
#include <hip/hip_bf16.h>

typedef __attribute__((ext_vector_type(16))) _Float16 v16h;
typedef __attribute__((ext_vector_type(8)))  _Float16 v8h;
typedef __attribute__((ext_vector_type(4)))  _Float16 v4h;
typedef __attribute__((ext_vector_type(8)))  float    v8f;

#define GN   50000          // nodes
#define GIN  64             // input dim
#define GD   128            // state dim
#define GH   8              // heads
#define GC   16             // per-head channels
#define GE   800000         // edges
#define GEP  (GE + GN)      // edges + self loops
#define GNL  4              // layers
#define GA   512            // action dim
#define GHID (GN / 4)       // 12500
#define NEG_GAT  0.2f
#define NEG_HEAD 0.01f
#define LN_EPS   1e-6f

// ---------------------------------------------------------------------------
// Float atomic max via CAS (all values finite; self-loops guarantee coverage)
// ---------------------------------------------------------------------------
__device__ __forceinline__ void atomicMaxF(float* addr, float val) {
    unsigned int* ua = (unsigned int*)addr;
    unsigned int old = *ua;
    while (true) {
        float f = __uint_as_float(old);
        if (f >= val) break;
        unsigned int assumed = old;
        old = atomicCAS(ua, assumed, __float_as_uint(val));
        if (old == assumed) break;
    }
}

// ---------------------------------------------------------------------------
// Fragment loaders.
// A (16x32 f16 layout, ISA 7.12.2): lane m=L&15, koff=(L>>4)*8:
//   a[0..7]=A[m][koff..koff+7], a[8..15]=A[m][16+koff..16+koff+7]
// B (32x16): lane n=L&15, kb=(L>>4)*8: split-8 order as A, b[0..7]=W[col][kb..kb+7], b[8..15]=W[col][16+kb..16+kb+7]
// ---------------------------------------------------------------------------
__device__ __forceinline__ v16h frag_split8_h(const _Float16* __restrict__ p) {
    v8h lo = *(const v8h*)(p);          // 16B -> b128
    v8h hi = *(const v8h*)(p + 16);
    return __builtin_shufflevector(lo, hi, 0,1,2,3,4,5,6,7,8,9,10,11,12,13,14,15);
}
__device__ __forceinline__ v16h frag_contig16_h(const _Float16* __restrict__ p) {
    v8h lo = *(const v8h*)(p);
    v8h hi = *(const v8h*)(p + 8);
    return __builtin_shufflevector(lo, hi, 0,1,2,3,4,5,6,7,8,9,10,11,12,13,14,15);
}
__device__ __forceinline__ v16h frag_split8_f(const float* __restrict__ p) {
    float4 p0 = *(const float4*)(p);
    float4 p1 = *(const float4*)(p + 4);
    float4 p2 = *(const float4*)(p + 16);
    float4 p3 = *(const float4*)(p + 20);
    v16h a;
    a[0]=(_Float16)p0.x; a[1]=(_Float16)p0.y; a[2]=(_Float16)p0.z; a[3]=(_Float16)p0.w;
    a[4]=(_Float16)p1.x; a[5]=(_Float16)p1.y; a[6]=(_Float16)p1.z; a[7]=(_Float16)p1.w;
    a[8]=(_Float16)p2.x; a[9]=(_Float16)p2.y; a[10]=(_Float16)p2.z; a[11]=(_Float16)p2.w;
    a[12]=(_Float16)p3.x; a[13]=(_Float16)p3.y; a[14]=(_Float16)p3.z; a[15]=(_Float16)p3.w;
    return a;
}

// ---------------------------------------------------------------------------
// WMMA GEMM: Y[50000,M] = act( X[.,KC] * W[M,KC]^T [+ bias] [+ res] )
// Wave = 5 consecutive 16-row tiles x one 16-col tile (3125 = 5*625 exact).
// B fragment reused across 5 wmma per K-step; all 5 A fragments loaded first,
// then 5 back-to-back v_wmma (disjoint regs -> no hazard nops between them).
// A_HALF: X is _Float16 (pre-converted); else f32 with inline cvt.
// OUT_F32: write f32 to Yf, else f16 to Yh (for tensors only consumed as A).
// ---------------------------------------------------------------------------
template<int KC, int ACT, bool HAS_BIAS, bool HAS_RES, bool A_HALF, bool OUT_F32>
__global__ void gemm_wmma_nt(const void* __restrict__ Xv,
                             const _Float16* __restrict__ W,
                             const float* __restrict__ bias,
                             const float* __restrict__ res,
                             float* __restrict__ Yf,
                             _Float16* __restrict__ Yh, int M) {
    constexpr int RPT = 5;
    const int lane = threadIdx.x;                    // 0..31
    const int ct   = blockIdx.y * 4 + threadIdx.y;   // col tile
    const int rt0  = blockIdx.x * RPT;               // first row tile

    const int mrow = lane & 15;
    const int koff = (lane >> 4) * 8;                // A K-subgroup
    const int kb   = (lane >> 4) * 8;                // B K-subgroup

    const _Float16* __restrict__ brow = W + (size_t)(ct * 16 + mrow) * KC;

    const _Float16* arow_h[RPT];
    const float*    arow_f[RPT];
#pragma unroll
    for (int t = 0; t < RPT; ++t) {
        const size_t ro = (size_t)((rt0 + t) * 16 + mrow) * KC;
        if (A_HALF) arow_h[t] = (const _Float16*)Xv + ro + koff;
        else        arow_f[t] = (const float*)Xv + ro + koff;
    }

    v8f acc[RPT];
#pragma unroll
    for (int t = 0; t < RPT; ++t) acc[t] = (v8f){};

#pragma unroll
    for (int k0 = 0; k0 < KC; k0 += 32) {
        v16h b = frag_split8_h(brow + k0 + kb);
        v16h a[RPT];
#pragma unroll
        for (int t = 0; t < RPT; ++t)
            a[t] = A_HALF ? frag_split8_h(arow_h[t] + k0)
                          : frag_split8_f(arow_f[t] + k0);
#pragma unroll
        for (int t = 0; t < RPT; ++t)
            acc[t] = __builtin_amdgcn_wmma_f32_16x16x32_f16(
                         false, a[t], false, b, (short)0, acc[t], false, false);
    }

    const int col = ct * 16 + (lane & 15);
    float bv = 0.f;
    if (HAS_BIAS) bv = bias[col];
#pragma unroll
    for (int t = 0; t < RPT; ++t) {
        const int rbase = (rt0 + t) * 16 + (lane >> 4) * 8;
#pragma unroll
        for (int r = 0; r < 8; ++r) {
            const size_t off = (size_t)(rbase + r) * M + col;
            float val = acc[t][r] + bv;
            if (HAS_RES) val += res[off];
            if (ACT == 1) val = fmaxf(val, 0.f);
            if (OUT_F32) Yf[off] = val;
            else         Yh[off] = (_Float16)val;
        }
    }
}

// ---------------------------------------------------------------------------
// f32 -> f16 conversion, 4 elements/thread (n multiple of 4)
// ---------------------------------------------------------------------------
__global__ void cvt_f32_f16(const float* __restrict__ src,
                            _Float16* __restrict__ dst, int n4) {
    int i = blockIdx.x * blockDim.x + threadIdx.x;
    if (i >= n4) return;
    float4 p = ((const float4*)src)[i];
    v4h o;
    o[0] = (_Float16)p.x; o[1] = (_Float16)p.y;
    o[2] = (_Float16)p.z; o[3] = (_Float16)p.w;
    *(v4h*)(dst + 4 * i) = o;
}

// ---------------------------------------------------------------------------
// Per-layer init: mx = -inf, den = 0, hnew[n,d] = gat_bias[d]
// ---------------------------------------------------------------------------
__global__ void gat_init(float* __restrict__ mx, float* __restrict__ den,
                         float* __restrict__ hn, const float* __restrict__ gb) {
    int idx = blockIdx.x * blockDim.x + threadIdx.x;
    if (idx < GN * GH) { mx[idx] = -3.402823466e38f; den[idx] = 0.f; }
    if (idx < GN * GD) hn[idx] = gb[idx & (GD - 1)];
}

// ---------------------------------------------------------------------------
// Edge pass 1: e = att . leaky_relu(xl[src]+xr[dst]); atomic max per (dst,h)
// ---------------------------------------------------------------------------
__global__ void gat_edge_logits(const float* __restrict__ xl,
                                const float* __restrict__ xr,
                                const float* __restrict__ att,
                                const int* __restrict__ ei,
                                float* __restrict__ eb,
                                float* __restrict__ mx) {
    int idx = blockIdx.x * blockDim.x + threadIdx.x;
    if (idx >= GEP * GH) return;
    const int eid = idx >> 3, hh = idx & 7;
    int s, t;
    if (eid < GE) { s = ei[eid]; t = ei[GE + eid]; } else { s = t = eid - GE; }
    const float4* pl = (const float4*)(xl + (size_t)s * GD + hh * GC);
    const float4* pr = (const float4*)(xr + (size_t)t * GD + hh * GC);
    const float4* pa = (const float4*)(att + hh * GC);
    float acc = 0.f;
#pragma unroll
    for (int i = 0; i < 4; ++i) {
        float4 l = pl[i], r = pr[i], a = pa[i];
        float u;
        u = l.x + r.x; u = (u > 0.f) ? u : NEG_GAT * u; acc += u * a.x;
        u = l.y + r.y; u = (u > 0.f) ? u : NEG_GAT * u; acc += u * a.y;
        u = l.z + r.z; u = (u > 0.f) ? u : NEG_GAT * u; acc += u * a.z;
        u = l.w + r.w; u = (u > 0.f) ? u : NEG_GAT * u; acc += u * a.w;
    }
    eb[idx] = acc;
    atomicMaxF(&mx[t * GH + hh], acc);
}

// ---------------------------------------------------------------------------
// Edge pass 2: ex = exp(e - mx[dst]); atomic add to den
// ---------------------------------------------------------------------------
__global__ void gat_edge_exp(float* __restrict__ eb,
                             const float* __restrict__ mx,
                             float* __restrict__ den,
                             const int* __restrict__ ei) {
    int idx = blockIdx.x * blockDim.x + threadIdx.x;
    if (idx >= GEP * GH) return;
    const int eid = idx >> 3, hh = idx & 7;
    int t;
    if (eid < GE) t = ei[GE + eid]; else t = eid - GE;
    float ex = __expf(eb[idx] - mx[t * GH + hh]);
    eb[idx] = ex;
    atomicAdd(&den[t * GH + hh], ex);
}

// ---------------------------------------------------------------------------
// Edge pass 3: hn[dst] += (ex/den[dst]) * xl[src]
// ---------------------------------------------------------------------------
__global__ void gat_edge_agg(const float* __restrict__ eb,
                             const float* __restrict__ den,
                             const float* __restrict__ xl,
                             const int* __restrict__ ei,
                             float* __restrict__ hn) {
    int idx = blockIdx.x * blockDim.x + threadIdx.x;
    if (idx >= GEP * GH) return;
    const int eid = idx >> 3, hh = idx & 7;
    int s, t;
    if (eid < GE) { s = ei[eid]; t = ei[GE + eid]; } else { s = t = eid - GE; }
    const float alpha = eb[idx] / den[t * GH + hh];
    const float4* pl = (const float4*)(xl + (size_t)s * GD + hh * GC);
    float* po = hn + (size_t)t * GD + hh * GC;
#pragma unroll
    for (int i = 0; i < 4; ++i) {
        float4 l = pl[i];
        atomicAdd(&po[4 * i + 0], alpha * l.x);
        atomicAdd(&po[4 * i + 1], alpha * l.y);
        atomicAdd(&po[4 * i + 2], alpha * l.z);
        atomicAdd(&po[4 * i + 3], alpha * l.w);
    }
}

// ---------------------------------------------------------------------------
// LayerNorm over D=128 + optional relu; writes f32 (head) and f16 (next GEMM)
// ---------------------------------------------------------------------------
__global__ void ln_kernel(const float* __restrict__ X,
                          const float* __restrict__ g,
                          const float* __restrict__ beta,
                          float* __restrict__ Y,
                          _Float16* __restrict__ Yh, int doRelu) {
    __shared__ float red[GD];
    const int row = blockIdx.x, tid = threadIdx.x;
    float val = X[(size_t)row * GD + tid];
    red[tid] = val; __syncthreads();
    for (int o = 64; o; o >>= 1) { if (tid < o) red[tid] += red[tid + o]; __syncthreads(); }
    float mu = red[0] * (1.f / GD); __syncthreads();
    float d = val - mu;
    red[tid] = d * d; __syncthreads();
    for (int o = 64; o; o >>= 1) { if (tid < o) red[tid] += red[tid + o]; __syncthreads(); }
    float var = red[0] * (1.f / GD);
    float y = d * rsqrtf(var + LN_EPS) * g[tid] + beta[tid];
    if (doRelu) y = fmaxf(y, 0.f);
    Y[(size_t)row * GD + tid]  = y;
    Yh[(size_t)row * GD + tid] = (_Float16)y;
}

// ---------------------------------------------------------------------------
// v[n] = dot(h[n,:], down_w) + down_b   (wave per node, 1 float4 per lane)
// ---------------------------------------------------------------------------
__global__ void down_kernel(const float* __restrict__ h,
                            const float* __restrict__ dw,
                            const float* __restrict__ db,
                            float* __restrict__ v) {
    const int wid  = (blockIdx.x * blockDim.x + threadIdx.x) >> 5;
    const int lane = threadIdx.x & 31;
    if (wid >= GN) return;
    float4 rv = ((const float4*)(h + (size_t)wid * GD))[lane];
    float4 wv = ((const float4*)dw)[lane];
    float acc = rv.x * wv.x + rv.y * wv.y + rv.z * wv.z + rv.w * wv.w;
    for (int o = 16; o; o >>= 1) acc += __shfl_down(acc, o);
    if (lane == 0) v[wid] = acc + db[0];
}

// ---------------------------------------------------------------------------
// Streaming matvec (the 2.5 GB out_w read dominates): block per row,
// 256 threads stride K, global_prefetch hint, shared reduce.
// ---------------------------------------------------------------------------
template<int LEAKY, int VEC>
__global__ void matvec_row(const float* __restrict__ vin,
                           const float* __restrict__ W,
                           const float* __restrict__ b,
                           float* __restrict__ y, int K) {
    __shared__ float red[256];
    const int r = blockIdx.x, tid = threadIdx.x;
    const float* __restrict__ row = W + (size_t)r * K;
    float acc = 0.f;
    if (VEC) {
        const float4* __restrict__ row4 = (const float4*)row;
        const float4* __restrict__ vin4 = (const float4*)vin;
        const int K4 = K >> 2;
        for (int j = tid; j < K4; j += 256) {
            __builtin_prefetch(&row4[j + 1024], 0, 1);   // global_prefetch_b8
            float4 w4 = row4[j], x4 = vin4[j];
            acc += w4.x * x4.x + w4.y * x4.y + w4.z * x4.z + w4.w * x4.w;
        }
    } else {
        for (int j = tid; j < K; j += 256) acc += vin[j] * row[j];
    }
    red[tid] = acc; __syncthreads();
    for (int o = 128; o; o >>= 1) { if (tid < o) red[tid] += red[tid + o]; __syncthreads(); }
    if (tid == 0) {
        float s = red[0] + b[r];
        if (LEAKY) s = (s > 0.f) ? s : NEG_HEAD * s;
        y[r] = s;
    }
}

// ---------------------------------------------------------------------------
extern "C" void kernel_launch(void* const* d_in, const int* in_sizes, int n_in,
                              void* d_out, int out_size, void* d_ws, size_t ws_size,
                              hipStream_t stream) {
    const float* x       = (const float*)d_in[0];
    const int*   ei      = (const int*)  d_in[1];
    const float* lin_w   = (const float*)d_in[2];
    const float* lin_b   = (const float*)d_in[3];
    const float* Wl      = (const float*)d_in[4];
    const float* Wr      = (const float*)d_in[5];
    const float* att     = (const float*)d_in[6];
    const float* gat_b   = (const float*)d_in[7];
    const float* w1      = (const float*)d_in[8];
    const float* b1      = (const float*)d_in[9];
    const float* w2      = (const float*)d_in[10];
    const float* b2      = (const float*)d_in[11];
    const float* ln_g    = (const float*)d_in[12];
    const float* ln_beta = (const float*)d_in[13];
    const float* down_w  = (const float*)d_in[14];
    const float* down_b  = (const float*)d_in[15];
    const float* out_w   = (const float*)d_in[16];
    const float* out_b   = (const float*)d_in[17];
    const float* out2_w  = (const float*)d_in[18];
    const float* out2_b  = (const float*)d_in[19];
    float* out = (float*)d_out;

    // ---- workspace partition: f32 region, then f16 region ----
    float* h   = (float*)d_ws;                 // [N,D] post-LN f32 (head input)
    float* xl  = h   + (size_t)GN * GD;        // [N,D]
    float* xr  = xl  + (size_t)GN * GD;        // [N,D] (also FFN out h2)
    float* hn  = xr  + (size_t)GN * GD;        // [N,D] GAT output / residual
    float* eb  = hn  + (size_t)GN * GD;        // [EP,H]
    float* mx  = eb  + (size_t)GEP * GH;       // [N,H]
    float* den = mx  + (size_t)GN * GH;        // [N,H]
    float* v   = den + (size_t)GN * GH;        // [N]
    float* y1  = v   + GN;                     // [HID]
    _Float16* h16   = (_Float16*)(y1 + GHID);  // [N,D]
    _Float16* x16   = h16  + (size_t)GN * GD;  // [N,IN]
    _Float16* t16   = x16  + (size_t)GN * GIN; // [N,D] FFN mid (f16 only)
    _Float16* lw16  = t16  + (size_t)GN * GD;  // lin_w  [D,IN]
    _Float16* Wl16  = lw16 + (size_t)GD * GIN; // [NL,D,D]
    _Float16* Wr16  = Wl16 + (size_t)GNL * GD * GD;
    _Float16* w116  = Wr16 + (size_t)GNL * GD * GD;
    _Float16* w216  = w116 + (size_t)GNL * GD * GD;

    const dim3 gblk(32, 4);
    const dim3 ggrd(625, GD / 64);             // 3125 row tiles / 5 per wave
    const int  egrid = (GEP * GH + 255) / 256;
    const int  igrid = (GN * GD + 255) / 256;
    auto cvt = [&](const float* s, _Float16* d, int n) {
        int n4 = n >> 2;
        cvt_f32_f16<<<(n4 + 255) / 256, 256, 0, stream>>>(s, d, n4);
    };

    // one-time f16 conversions (weights tiny; x = 6.4 MB)
    cvt(x, x16, GN * GIN);
    cvt(lin_w, lw16, GD * GIN);
    cvt(Wl, Wl16, GNL * GD * GD);
    cvt(Wr, Wr16, GNL * GD * GD);
    cvt(w1, w116, GNL * GD * GD);
    cvt(w2, w216, GNL * GD * GD);

    // input projection: h16 = f16( x @ lin_w.T + lin_b )   (only used as A)
    gemm_wmma_nt<GIN, 0, true, false, true, false>
        <<<ggrd, gblk, 0, stream>>>(x16, lw16, lin_b, nullptr, nullptr, h16, GD);

    for (int l = 0; l < GNL; ++l) {
        // xl = h @ Wl.T ; xr = h @ Wr.T  (A = h16, out f32 for edge kernels)
        gemm_wmma_nt<GD, 0, false, false, true, true>
            <<<ggrd, gblk, 0, stream>>>(h16, Wl16 + (size_t)l * GD * GD,
                                        nullptr, nullptr, xl, nullptr, GD);
        gemm_wmma_nt<GD, 0, false, false, true, true>
            <<<ggrd, gblk, 0, stream>>>(h16, Wr16 + (size_t)l * GD * GD,
                                        nullptr, nullptr, xr, nullptr, GD);
        // init mx/den + seed hn with gat bias
        gat_init<<<igrid, 256, 0, stream>>>(mx, den, hn, gat_b + (size_t)l * GD);
        // segment softmax + aggregate
        gat_edge_logits<<<egrid, 256, 0, stream>>>(xl, xr, att + (size_t)l * GH * GC, ei, eb, mx);
        gat_edge_exp  <<<egrid, 256, 0, stream>>>(eb, mx, den, ei);
        gat_edge_agg  <<<egrid, 256, 0, stream>>>(eb, den, xl, ei, hn);
        // FFN: t16 = f16(relu(hn @ w1.T + b1)) ; h2(xr) = t16 @ w2.T + b2 + hn
        gemm_wmma_nt<GD, 1, true, false, false, false>
            <<<ggrd, gblk, 0, stream>>>(hn, w116 + (size_t)l * GD * GD,
                                        b1 + (size_t)l * GD, nullptr, nullptr, t16, GD);
        gemm_wmma_nt<GD, 0, true, true, true, true>
            <<<ggrd, gblk, 0, stream>>>(t16, w216 + (size_t)l * GD * GD,
                                        b2 + (size_t)l * GD, hn, xr, nullptr, GD);
        // h, h16 = LN(h2) (+relu except last layer)
        ln_kernel<<<GN, GD, 0, stream>>>(xr, ln_g + (size_t)l * GD,
                                         ln_beta + (size_t)l * GD, h, h16,
                                         (l < GNL - 1) ? 1 : 0);
    }

    // head: v = h @ down_w.T + down_b ; y1 = leaky(out_w v + out_b) ; out = out2_w y1 + out2_b
    down_kernel<<<(GN * 32 + 255) / 256, 256, 0, stream>>>(h, down_w, down_b, v);
    matvec_row<1, 1><<<GHID, 256, 0, stream>>>(v,  out_w,  out_b,  y1,  GN);
    matvec_row<0, 0><<<GA,   256, 0, stream>>>(y1, out2_w, out2_b, out, GHID);
}
